// VQCodebook_14585708937328
// MI455X (gfx1250) — matrix operand order repaired and checked by reading, not compiled, not run on hardware
//
#include <hip/hip_runtime.h>

typedef __attribute__((ext_vector_type(16))) __bf16 v16bf;
typedef __attribute__((ext_vector_type(8)))  float  v8f;

#define N_CODES   1024
#define CODE_DIM  64
#define M_TOKENS  (16 * 8192)
#define CHUNK     128
#define NCHUNKS   (N_CODES / CHUNK)
#define LDS_PITCH 72   // 64 + 8 bf16 pad -> 144B row stride, kills DS bank conflicts

// ---------- bf16 split helpers (round-to-nearest-even) ----------
__device__ __forceinline__ __bf16 f2bf(float f) {
  unsigned u = __builtin_bit_cast(unsigned, f);
  unsigned r = u + 0x7FFFu + ((u >> 16) & 1u);
  unsigned short s = (unsigned short)(r >> 16);
  return __builtin_bit_cast(__bf16, s);
}
__device__ __forceinline__ float bf2f(__bf16 b) {
  unsigned short s = __builtin_bit_cast(unsigned short, b);
  unsigned u = ((unsigned)s) << 16;
  return __builtin_bit_cast(float, u);
}

// async 16B global->LDS copy (GLOBAL_LOAD_ASYNC_TO_LDS_B128, ASYNCcnt-tracked).
// ldsAddr: low 32 bits of a generic pointer into __shared__ (== DS byte offset).
__device__ __forceinline__ void async_copy_b128(unsigned ldsAddr, const void* gsrc) {
  asm volatile("global_load_async_to_lds_b128 %0, %1, off"
               :: "v"(ldsAddr), "v"((unsigned long long)(uintptr_t)gsrc)
               : "memory");
}
__device__ __forceinline__ void wait_async_zero() {
  asm volatile("s_wait_asynccnt 0x0" ::: "memory");
}

struct B256 { uint4 x, y; };  // 32B container for one v16bf

// ---------- kernel 1: split codebook into bf16 hi/lo + norms ----------
__global__ __launch_bounds__(256) void vq_prep(const float* __restrict__ cb,
                                               unsigned short* __restrict__ cbHi,
                                               unsigned short* __restrict__ cbLo,
                                               float* __restrict__ norms) {
  int n = blockIdx.x * 256 + threadIdx.x;
  if (n >= N_CODES) return;
  const float* row = cb + (size_t)n * CODE_DIM;
  float s = 0.f;
  #pragma unroll
  for (int d = 0; d < CODE_DIM; ++d) {
    float v = row[d];
    s += v * v;
    __bf16 hi = f2bf(v);
    __bf16 lo = f2bf(v - bf2f(hi));
    cbHi[(size_t)n * CODE_DIM + d] = __builtin_bit_cast(unsigned short, hi);
    cbLo[(size_t)n * CODE_DIM + d] = __builtin_bit_cast(unsigned short, lo);
  }
  norms[n] = s;
}

// ---------- kernel 2: WMMA distance GEMM + argmin + gather + loss partials ----------
__global__ __launch_bounds__(256) void vq_main(const float* __restrict__ z,
                                               const float* __restrict__ cb,
                                               const unsigned short* __restrict__ cbHi,
                                               const unsigned short* __restrict__ cbLo,
                                               const float* __restrict__ norms,
                                               float* __restrict__ outZ,
                                               float* __restrict__ outIdx,
                                               float* __restrict__ partials) {
  __shared__ unsigned short sHi[CHUNK][LDS_PITCH];
  __shared__ unsigned short sLo[CHUNK][LDS_PITCH];
  __shared__ float sNorm[CHUNK];
  __shared__ float sWred[8];

  const int tid  = threadIdx.x;
  const int wave = tid >> 5;
  const int lane = tid & 31;
  const int h    = (lane >> 4) & 1;   // which 16-lane half
  const int nsub = lane & 15;

  const int m0 = blockIdx.x * 128 + wave * 16;   // 16-token tile per wave

  // ---- A tile: z rows, split to bf16 hi/lo, ISA 16x32 A layout ----
  // lane half h, v16bf elems 0..7  -> K = kk*32 + h*8 + e
  //             elems 8..15 -> K = kk*32 + 16 + h*8 + (e-8)
  v16bf aHi[2], aLo[2];
  {
    const float* zr = z + (size_t)(m0 + nsub) * CODE_DIM;
    #pragma unroll
    for (int kk = 0; kk < 2; ++kk) {
      #pragma unroll
      for (int c = 0; c < 2; ++c) {
        const int d0 = kk * 32 + c * 16 + h * 8;
        float4 f0 = *(const float4*)(zr + d0);
        float4 f1 = *(const float4*)(zr + d0 + 4);
        float v[8] = {f0.x, f0.y, f0.z, f0.w, f1.x, f1.y, f1.z, f1.w};
        #pragma unroll
        for (int e = 0; e < 8; ++e) {
          __bf16 hi = f2bf(v[e]);
          __bf16 lo = f2bf(v[e] - bf2f(hi));
          aHi[kk][c * 8 + e] = hi;
          aLo[kk][c * 8 + e] = lo;
        }
      }
    }
  }

  float bestVal[8];
  int   bestIdx[8];
  #pragma unroll
  for (int r = 0; r < 8; ++r) { bestVal[r] = 3.0e38f; bestIdx[r] = 0; }

  #pragma unroll 1
  for (int c = 0; c < NCHUNKS; ++c) {
    const int nBase = c * CHUNK;
    // stage 128 codes (hi+lo) into LDS with async global->LDS b128 copies
    const uint4* gHi = (const uint4*)(cbHi + (size_t)nBase * CODE_DIM);
    const uint4* gLo = (const uint4*)(cbLo + (size_t)nBase * CODE_DIM);
    #pragma unroll
    for (int t = 0; t < 4; ++t) {
      int i = tid + t * 256;              // 1024 uint4 per array
      int row = i >> 3, q = i & 7;
      async_copy_b128((unsigned)(uintptr_t)&sHi[row][q * 8], gHi + i);
      async_copy_b128((unsigned)(uintptr_t)&sLo[row][q * 8], gLo + i);
    }
    if (tid < CHUNK) sNorm[tid] = norms[nBase + tid];
    if ((c + 1 < NCHUNKS) && tid < CHUNK) {   // pull next chunk toward L2/L0
      __builtin_prefetch(cbHi + (size_t)(nBase + CHUNK) * CODE_DIM + tid * 64, 0, 0);
      __builtin_prefetch(cbLo + (size_t)(nBase + CHUNK) * CODE_DIM + tid * 64, 0, 0);
    }
    wait_async_zero();    // our async copies landed in LDS
    __syncthreads();      // ... and everyone else's too

    #pragma unroll 2
    for (int t = 0; t < CHUNK / 16; ++t) {
      const int nl = t * 16 + nsub;       // this lane's code column within chunk
      // B 32x16 wave32 layout: lane half h holds K kk*32+h*8+[0,8) then kk*32+16+h*8+[0,8)
      v16bf bHi[2], bLo[2];
      #pragma unroll
      for (int kk = 0; kk < 2; ++kk) {
        const unsigned short* pH = &sHi[nl][kk * 32 + h * 8];
        const unsigned short* pL = &sLo[nl][kk * 32 + h * 8];
        B256 th; th.x = *(const uint4*)pH; th.y = *(const uint4*)(pH + 16);
        B256 tl; tl.x = *(const uint4*)pL; tl.y = *(const uint4*)(pL + 16);
        bHi[kk] = __builtin_bit_cast(v16bf, th);
        bLo[kk] = __builtin_bit_cast(v16bf, tl);
      }
      // split-precision dot: zh*eh + zh*el + zl*eh  (error ~2^-16 relative)
      v8f acc = {0.f, 0.f, 0.f, 0.f, 0.f, 0.f, 0.f, 0.f};
      acc = __builtin_amdgcn_wmma_f32_16x16x32_bf16(false, aHi[0], false, bHi[0], (short)0, acc, false, false);
      acc = __builtin_amdgcn_wmma_f32_16x16x32_bf16(false, aHi[1], false, bHi[1], (short)0, acc, false, false);
      acc = __builtin_amdgcn_wmma_f32_16x16x32_bf16(false, aHi[0], false, bLo[0], (short)0, acc, false, false);
      acc = __builtin_amdgcn_wmma_f32_16x16x32_bf16(false, aHi[1], false, bLo[1], (short)0, acc, false, false);
      acc = __builtin_amdgcn_wmma_f32_16x16x32_bf16(false, aLo[0], false, bHi[0], (short)0, acc, false, false);
      acc = __builtin_amdgcn_wmma_f32_16x16x32_bf16(false, aLo[1], false, bHi[1], (short)0, acc, false, false);

      const float nn     = sNorm[nl];
      const int codeIdx  = nBase + nl;
      #pragma unroll
      for (int r = 0; r < 8; ++r) {
        // ||e||^2 - 2 z.e  (||z||^2 constant per token -> argmin-invariant)
        float d2 = fmaf(-2.0f, acc[r], nn);
        if (d2 < bestVal[r]) { bestVal[r] = d2; bestIdx[r] = codeIdx; }
      }
    }
    __syncthreads();
  }

  // argmin across the 16 lanes of each half (rows r+8h live per half)
  #pragma unroll
  for (int m = 8; m >= 1; m >>= 1) {
    #pragma unroll
    for (int r = 0; r < 8; ++r) {
      float ov = __shfl_xor(bestVal[r], m, 32);
      int   oi = __shfl_xor(bestIdx[r], m, 32);
      if (ov < bestVal[r] || (ov == bestVal[r] && oi < bestIdx[r])) {
        bestVal[r] = ov; bestIdx[r] = oi;
      }
    }
  }

  // gather z_q (f32 codebook), write outputs, accumulate loss partial
  float lsum = 0.f;
  #pragma unroll
  for (int r = 0; r < 8; ++r) {
    const int row = r + 8 * h;
    const size_t token = (size_t)m0 + row;
    const float4 c4 = *(const float4*)(cb + (size_t)bestIdx[r] * CODE_DIM + nsub * 4);
    const float4 z4 = *(const float4*)(z + token * CODE_DIM + nsub * 4);
    float dx = c4.x - z4.x, dy = c4.y - z4.y, dz = c4.z - z4.z, dw = c4.w - z4.w;
    lsum += dx * dx + dy * dy + dz * dz + dw * dw;
    *(float4*)(outZ + token * CODE_DIM + nsub * 4) = c4;   // z_q_st forward == z_q
    if (nsub == r) outIdx[token] = (float)bestIdx[r];
  }
  #pragma unroll
  for (int m = 16; m >= 1; m >>= 1) lsum += __shfl_xor(lsum, m, 32);
  if (lane == 0) sWred[wave] = lsum;
  __syncthreads();
  if (tid == 0) {
    float s = 0.f;
    #pragma unroll
    for (int w = 0; w < 8; ++w) s += sWred[w];
    partials[blockIdx.x] = s;   // fixed-order => deterministic
  }
}

// ---------- kernel 3: deterministic final loss reduction ----------
__global__ __launch_bounds__(256) void vq_loss_reduce(const float* __restrict__ partials,
                                                      float* __restrict__ loss) {
  __shared__ float s[256];
  float a = 0.f;
  for (int i = threadIdx.x; i < M_TOKENS / 128; i += 256) a += partials[i];
  s[threadIdx.x] = a;
  __syncthreads();
  for (int off = 128; off > 0; off >>= 1) {
    if (threadIdx.x < off) s[threadIdx.x] += s[threadIdx.x + off];
    __syncthreads();
  }
  // vq_loss = (1 + COMMITMENT) * mean((z_q - z)^2), COMMITMENT = 0.25
  if (threadIdx.x == 0) *loss = s[0] * (1.25f / (float)((size_t)M_TOKENS * CODE_DIM));
}

extern "C" void kernel_launch(void* const* d_in, const int* in_sizes, int n_in,
                              void* d_out, int out_size, void* d_ws, size_t ws_size,
                              hipStream_t stream) {
  (void)in_sizes; (void)n_in; (void)out_size; (void)ws_size;
  const float* z  = (const float*)d_in[0];   // [16*8192, 64] f32
  const float* cb = (const float*)d_in[1];   // [1024, 64] f32

  // workspace layout (bytes): cbHi 128K | cbLo 128K | norms 4K | partials 4K
  unsigned short* wsHi = (unsigned short*)d_ws;
  unsigned short* wsLo = wsHi + (size_t)N_CODES * CODE_DIM;
  float* norms    = (float*)((char*)d_ws + 2u * N_CODES * CODE_DIM * sizeof(unsigned short));
  float* partials = norms + N_CODES;

  float* outZ    = (float*)d_out;                          // [M,64] z_q_st
  float* outIdx  = outZ + (size_t)M_TOKENS * CODE_DIM;     // [M] indices (as float)
  float* outLoss = outIdx + M_TOKENS;                      // scalar

  vq_prep<<<N_CODES / 256, 256, 0, stream>>>(cb, wsHi, wsLo, norms);
  vq_main<<<M_TOKENS / 128, 256, 0, stream>>>(z, cb, wsHi, wsLo, norms, outZ, outIdx, partials);
  vq_loss_reduce<<<1, 256, 0, stream>>>(partials, outLoss);
}
